// SinkhornNetwork_30648886624803
// MI455X (gfx1250) — matrix operand order repaired and checked by reading, not compiled, not run on hardware
//
#include <hip/hip_runtime.h>
#include <hip/hip_bf16.h>
#include <cstdint>
#include <cstddef>

// ---------------------------------------------------------------------------
// SinkhornNetwork on MI455X (gfx1250, wave32, WMMA).
// GEMMs: bf16 v_wmma_f32_16x16x32_bf16, LDS-staged via async global->LDS
// copies (ASYNCcnt path), double-buffered. Sinkhorn in fp32.
// ---------------------------------------------------------------------------

typedef __attribute__((ext_vector_type(16))) __bf16 v16bf;
typedef __attribute__((ext_vector_type(8)))  __bf16 v8bf;
typedef __attribute__((ext_vector_type(8)))  float  v8f;

#define MROWS   18432         // B*N = 512*36
#define SEQ_D   3076
#define FEAT_LD 416           // 388 padded to multiple of 32
#define TAU_INV 20.0f
#define LDS_LDK 40            // 32 K-halves padded to 40 (bank-conflict-free)

// ---------------- async global -> LDS staging (ASYNCcnt) --------------------
__device__ __forceinline__ void async_b128(unsigned lds_off, const __bf16* gptr) {
  asm volatile("global_load_async_to_lds_b128 %0, %1, off"
               :: "v"(lds_off), "v"(gptr) : "memory");
}

// Stage a 128x32 A-tile and a 128x32 B-tile (bf16) for one K-step.
// 512 16B chunks per tile, 256 threads -> 2 chunks each; 4 async ops / wave.
__device__ __forceinline__ void stage_tile_pair(
    const __bf16* __restrict__ X, int ldx, int mb,
    const __bf16* __restrict__ W, int ldw, int nb,
    int k0, __bf16* At, __bf16* Bt, int tid) {
#pragma unroll
  for (int q = 0; q < 2; ++q) {
    const int c   = tid * 2 + q;     // 0..511
    const int row = c >> 2;          // 0..127
    const int kc  = (c & 3) * 8;     // 0,8,16,24 halves
    async_b128((unsigned)(size_t)(At + row * LDS_LDK + kc),
               X + (size_t)(mb + row) * ldx + k0 + kc);
    async_b128((unsigned)(size_t)(Bt + row * LDS_LDK + kc),
               W + (size_t)(nb + row) * ldw + k0 + kc);
  }
}

// ---------------- fragment loaders from LDS (ISA 7.12.2 bf16 layouts) -------
// A (16x32 MxK): lane L holds M=L%16; halves e: K=(e<8?0:16)+(L<16?0:8)+(e%8).
__device__ __forceinline__ v16bf lds_a_frag(const __bf16* Ab, int row0, int lane) {
  const __bf16* p = Ab + (row0 + (lane & 15)) * LDS_LDK + ((lane & 16) ? 8 : 0);
  v8bf lo = *(const v8bf*)(p);
  v8bf hi = *(const v8bf*)(p + 16);
  return __builtin_shufflevector(lo, hi, 0,1,2,3,4,5,6,7,8,9,10,11,12,13,14,15);
}
// B (32x16 KxN) from row-major W(N x K): lane L holds N=L%16; K=(e<8?0:16)+(L<16?0:8)+(e%8).
__device__ __forceinline__ v16bf lds_b_frag(const __bf16* Bb, int n0, int lane) {
  const __bf16* p = Bb + (n0 + (lane & 15)) * LDS_LDK + ((lane & 16) ? 8 : 0);
  v8bf lo = *(const v8bf*)(p);
  v8bf hi = *(const v8bf*)(p + 16);
  return __builtin_shufflevector(lo, hi, 0,1,2,3,4,5,6,7,8,9,10,11,12,13,14,15);
}

#define WMMA_BF16(A, B, C) \
  __builtin_amdgcn_wmma_f32_16x16x32_bf16(false, (A), false, (B), (short)0, (C), false, false)

// ---------------- fused GEMM: out = epilogue(X @ W^T + bias) ----------------
// Block tile 128x128, 8 waves, wave tile 32x64 (8 WMMA / K-step).
// EPI==0: relu -> bf16 at column offset ncol_off. EPI==1: tanh -> fp32, col<nvalid.
template <int EPI>
__global__ __launch_bounds__(256)
void gemm_wmma_bf16(const __bf16* __restrict__ X, int ldx,
                    const __bf16* __restrict__ W, int ldw,
                    const float* __restrict__ bias,
                    void* __restrict__ outp, int ldo, int ncol_off, int nvalid,
                    int K) {
  __shared__ __bf16 At[2][128 * LDS_LDK];
  __shared__ __bf16 Bt[2][128 * LDS_LDK];

  const int tid  = threadIdx.x;
  const int lane = tid & 31;
  const int wave = tid >> 5;                 // 8 waves
  const int mb   = blockIdx.x * 128;
  const int nb   = blockIdx.y * 128;
  const int wm   = (wave & 3) * 32;          // wave tile 32(M) x 64(N)
  const int wn   = (wave >> 2) * 64;

  const int KT = K >> 5;
  stage_tile_pair(X, ldx, mb, W, ldw, nb, 0, At[0], Bt[0], tid);

  v8f acc[2][4] = {};
  for (int kt = 0; kt < KT; ++kt) {
    const int cur = kt & 1;
    if (kt + 1 < KT) {
      stage_tile_pair(X, ldx, mb, W, ldw, nb, (kt + 1) << 5, At[cur ^ 1], Bt[cur ^ 1], tid);
      asm volatile("s_wait_asynccnt 0x4" ::: "memory");   // current tile done
    } else {
      asm volatile("s_wait_asynccnt 0x0" ::: "memory");
    }
    __syncthreads();

    const __bf16* Ab = At[cur];
    const __bf16* Bb = Bt[cur];
    v16bf a0 = lds_a_frag(Ab, wm,      lane);
    v16bf a1 = lds_a_frag(Ab, wm + 16, lane);
    v16bf b0 = lds_b_frag(Bb, wn,      lane);
    v16bf b1 = lds_b_frag(Bb, wn + 16, lane);
    v16bf b2 = lds_b_frag(Bb, wn + 32, lane);
    v16bf b3 = lds_b_frag(Bb, wn + 48, lane);
    acc[0][0] = WMMA_BF16(a0, b0, acc[0][0]);
    acc[0][1] = WMMA_BF16(a0, b1, acc[0][1]);
    acc[0][2] = WMMA_BF16(a0, b2, acc[0][2]);
    acc[0][3] = WMMA_BF16(a0, b3, acc[0][3]);
    acc[1][0] = WMMA_BF16(a1, b0, acc[1][0]);
    acc[1][1] = WMMA_BF16(a1, b1, acc[1][1]);
    acc[1][2] = WMMA_BF16(a1, b2, acc[1][2]);
    acc[1][3] = WMMA_BF16(a1, b3, acc[1][3]);

    __syncthreads();   // all waves done reading buf[cur] before it is re-staged
  }

  // C/D layout: VGPR r, lanes 0-15 -> M=r, lanes 16-31 -> M=8+r; N = lane%16.
#pragma unroll
  for (int i = 0; i < 2; ++i) {
#pragma unroll
    for (int j = 0; j < 4; ++j) {
      const int rbase = mb + wm + 16 * i + ((lane & 16) ? 8 : 0);
      const int col   = nb + wn + 16 * j + (lane & 15);
      if (EPI == 0) {
        const float bv = bias[col];
        __bf16* o = (__bf16*)outp;
#pragma unroll
        for (int r = 0; r < 8; ++r) {
          float v = fmaxf(acc[i][j][r] + bv, 0.0f);
          o[(size_t)(rbase + r) * ldo + ncol_off + col] = (__bf16)v;
        }
      } else {
        if (col < nvalid) {
          const float bv = bias[col];
          float* o = (float*)outp;
#pragma unroll
          for (int r = 0; r < 8; ++r) {
            float v = tanhf(acc[i][j][r] + bv);
            o[(size_t)(rbase + r) * ldo + col] = v;
          }
        }
      }
    }
  }
}

// ---------------- seq split: fp32 -> packed bf16 branch inputs --------------
__global__ void split_seq_kernel(const float* __restrict__ seq,
                                 __bf16* __restrict__ xtxt,
                                 __bf16* __restrict__ xvis,
                                 __bf16* __restrict__ xsen) {
  size_t idx = (size_t)blockIdx.x * blockDim.x + threadIdx.x;   // MROWS*SEQ_D
  size_t row = idx / SEQ_D;
  int    c   = (int)(idx % SEQ_D);
  float  v   = seq[idx];
  if (c < 512)        xtxt[row * 512  + c]          = (__bf16)v;
  else if (c < 2560)  xvis[row * 2048 + (c - 512)]  = (__bf16)v;
  else if (c >= 2564) xsen[row * 512  + (c - 2564)] = (__bf16)v;
}

// ---------------- weight fp32 -> bf16 with zero padding ---------------------
__global__ void conv_weight_kernel(const float* __restrict__ src, __bf16* __restrict__ dst,
                                   int N, int K, int Npad, int Kpad) {
  size_t idx = (size_t)blockIdx.x * blockDim.x + threadIdx.x;
  if (idx >= (size_t)Npad * Kpad) return;
  int n = (int)(idx / Kpad), k = (int)(idx % Kpad);
  float v = (n < N && k < K) ? src[(size_t)n * K + k] : 0.0f;
  dst[idx] = (__bf16)v;
}

// ---------------- pos branch: relu(pos) into feat cols 256..259, zero pad ---
__global__ void pos_pad_kernel(const float* __restrict__ seq, __bf16* __restrict__ feat) {
  const int m = blockIdx.x;           // MROWS blocks
  const int j = threadIdx.x;          // 32 threads
  if (j < 4) {
    float v = fmaxf(seq[(size_t)m * SEQ_D + 2560 + j], 0.0f);
    feat[(size_t)m * FEAT_LD + 256 + j] = (__bf16)v;
  } else {
    feat[(size_t)m * FEAT_LD + 384 + j] = (__bf16)0.0f;   // cols 388..415 = 0
  }
}

// ---------------- sinkhorn: per batch element 36x36, 20 iterations ----------
__global__ __launch_bounds__(128)
void sinkhorn_kernel(const float* __restrict__ logits, float* __restrict__ out) {
  __shared__ float x[36][37];
  const int b   = blockIdx.x;
  const int tid = threadIdx.x;

  for (int idx = tid; idx < 36 * 36; idx += blockDim.x) {
    int r = idx / 36, c = idx % 36;
    x[r][c] = __expf(logits[((size_t)b * 36 + r) * 36 + c] * TAU_INV);
  }
  __syncthreads();

  for (int it = 0; it < 20; ++it) {
    if (tid < 36) {                          // rows: logsumexp over last axis
      float m = -1e30f;
      for (int c = 0; c < 36; ++c) m = fmaxf(m, x[tid][c]);
      float s = 0.0f;
      for (int c = 0; c < 36; ++c) s += __expf(x[tid][c] - m);
      float lse = m + __logf(s);
      for (int c = 0; c < 36; ++c) x[tid][c] -= lse;
    }
    __syncthreads();
    if (tid < 36) {                          // cols: logsumexp over axis -2
      float m = -1e30f;
      for (int r = 0; r < 36; ++r) m = fmaxf(m, x[r][tid]);
      float s = 0.0f;
      for (int r = 0; r < 36; ++r) s += __expf(x[r][tid] - m);
      float lse = m + __logf(s);
      for (int r = 0; r < 36; ++r) x[r][tid] -= lse;
    }
    __syncthreads();
  }

  for (int idx = tid; idx < 36 * 36; idx += blockDim.x) {
    int r = idx / 36, c = idx % 36;
    out[((size_t)b * 36 + r) * 36 + c] = __expf(x[r][c]);
  }
}

// ---------------------------------------------------------------------------
extern "C" void kernel_launch(void* const* d_in, const int* in_sizes, int n_in,
                              void* d_out, int out_size, void* d_ws, size_t ws_size,
                              hipStream_t stream) {
  const float* seq   = (const float*)d_in[0];
  const float* Wtxt  = (const float*)d_in[1];
  const float* Btxt  = (const float*)d_in[2];
  const float* Wvis1 = (const float*)d_in[3];
  const float* Bvis1 = (const float*)d_in[4];
  const float* Wvis2 = (const float*)d_in[5];
  const float* Bvis2 = (const float*)d_in[6];
  const float* Wsen  = (const float*)d_in[7];
  const float* Bsen  = (const float*)d_in[8];
  const float* Wfcp  = (const float*)d_in[9];
  const float* Bfcp  = (const float*)d_in[10];
  const float* Wfc   = (const float*)d_in[11];
  const float* Bfc   = (const float*)d_in[12];

  // workspace carve-up (256B aligned)
  char* base = (char*)d_ws;
  size_t off = 0;
  auto take = [&](size_t bytes) -> void* {
    void* p = base + off;
    off = (off + bytes + 255) & ~(size_t)255;
    return p;
  };
  __bf16* dXtxt  = (__bf16*)take((size_t)MROWS * 512  * 2);
  __bf16* dXvis  = (__bf16*)take((size_t)MROWS * 2048 * 2);
  __bf16* dXsen  = (__bf16*)take((size_t)MROWS * 512  * 2);
  __bf16* dHvis  = (__bf16*)take((size_t)MROWS * 512  * 2);
  __bf16* dFeat  = (__bf16*)take((size_t)MROWS * FEAT_LD * 2);
  __bf16* dHfcp  = (__bf16*)take((size_t)MROWS * 256  * 2);
  float*  dLogit = (float*) take((size_t)MROWS * 36   * 4);
  __bf16* dWtxt  = (__bf16*)take((size_t)128 * 512  * 2);
  __bf16* dWvis1 = (__bf16*)take((size_t)512 * 2048 * 2);
  __bf16* dWvis2 = (__bf16*)take((size_t)128 * 512  * 2);
  __bf16* dWsen  = (__bf16*)take((size_t)128 * 512  * 2);
  __bf16* dWfcp  = (__bf16*)take((size_t)256 * FEAT_LD * 2);
  __bf16* dWfc   = (__bf16*)take((size_t)128 * 256  * 2);  // N 36 -> 128 pad

  // 1) split + bf16 convert the input sequence
  {
    size_t total = (size_t)MROWS * SEQ_D;
    split_seq_kernel<<<dim3((unsigned)(total / 256)), 256, 0, stream>>>(seq, dXtxt, dXvis, dXsen);
  }
  // 2) convert weights (with padding where needed)
  auto convw = [&](const float* s, __bf16* d, int N, int K, int Npad, int Kpad) {
    size_t total = (size_t)Npad * Kpad;
    conv_weight_kernel<<<dim3((unsigned)((total + 255) / 256)), 256, 0, stream>>>(s, d, N, K, Npad, Kpad);
  };
  convw(Wtxt,  dWtxt,  128, 512,  128, 512);
  convw(Wvis1, dWvis1, 512, 2048, 512, 2048);
  convw(Wvis2, dWvis2, 128, 512,  128, 512);
  convw(Wsen,  dWsen,  128, 512,  128, 512);
  convw(Wfcp,  dWfcp,  256, 388,  256, FEAT_LD);
  convw(Wfc,   dWfc,   36,  256,  128, 256);

  const unsigned MB = MROWS / 128;   // 144 row-blocks
  // 3) vis1: (M x 2048) @ (2048 x 512) -> relu -> bf16
  gemm_wmma_bf16<0><<<dim3(MB, 512 / 128), 256, 0, stream>>>(
      dXvis, 2048, dWvis1, 2048, Bvis1, dHvis, 512, 0, 512, 2048);
  // 4) txt / vis2 / sen: (M x 512) @ (512 x 128) -> relu -> feat columns
  gemm_wmma_bf16<0><<<dim3(MB, 1), 256, 0, stream>>>(
      dXtxt, 512, dWtxt, 512, Btxt, dFeat, FEAT_LD, 0, 128, 512);
  gemm_wmma_bf16<0><<<dim3(MB, 1), 256, 0, stream>>>(
      dHvis, 512, dWvis2, 512, Bvis2, dFeat, FEAT_LD, 128, 128, 512);
  gemm_wmma_bf16<0><<<dim3(MB, 1), 256, 0, stream>>>(
      dXsen, 512, dWsen, 512, Bsen, dFeat, FEAT_LD, 260, 128, 512);
  // 5) pos branch + zero the K-padding of feat
  pos_pad_kernel<<<dim3(MROWS), 32, 0, stream>>>(seq, dFeat);
  // 6) fc_pos: (M x 416) @ (416 x 256) -> relu
  gemm_wmma_bf16<0><<<dim3(MB, 256 / 128), 256, 0, stream>>>(
      dFeat, FEAT_LD, dWfcp, FEAT_LD, Bfcp, dHfcp, 256, 0, 256, FEAT_LD);
  // 7) fc: (M x 256) @ (256 x 36pad128) -> tanh -> fp32 logits
  gemm_wmma_bf16<1><<<dim3(MB, 1), 256, 0, stream>>>(
      dHfcp, 256, dWfc, 256, Bfc, dLogit, 36, 0, 36, 256);
  // 8) sinkhorn normalization, one block per batch element
  sinkhorn_kernel<<<dim3(512), 128, 0, stream>>>(dLogit, (float*)d_out);

  (void)in_sizes; (void)n_in; (void)out_size; (void)ws_size;
}
